// MultiheadAttention_75290776699207
// MI455X (gfx1250) — compile-verified, matrix operand order checked by reading, not run on hardware
//
#include <hip/hip_runtime.h>
#include <hip/hip_bf16.h>

typedef _Float16 half_t;
typedef __attribute__((ext_vector_type(16))) _Float16 v16h;
typedef __attribute__((ext_vector_type(8)))  _Float16 v8h;
typedef __attribute__((ext_vector_type(8)))  float    v8f;
typedef __attribute__((ext_vector_type(4)))  float    v4f;

#define EMBED   1024
#define NHEADS  16
#define HDIM    64
#define SEQ     2048
#define BSZ     2
#define ROWS    (SEQ * BSZ)      // 4096 GEMM rows
#define SROW    2052             // padded LDS score row stride (floats); 8*SROW%64 != 0
#define FMIN    (-3.402823466e38f)

__device__ __forceinline__ v8f wmma_f16(v16h a, v16h b, v8f c) {
    return __builtin_amdgcn_wmma_f32_16x16x32_f16(false, a, false, b, (short)0, c,
                                                  false, false);
}

// Build a 16-half fragment from K-halves at p (K = base+8*hi .. +7) and p+16
// (K = base+16+8*hi .. +7), per the CDNA5 16-bit A/B VGPR layout.
__device__ __forceinline__ v16h frag_f16(const half_t* p) {
    const v8h lo = *(const v8h*)p;
    const v8h hi = *(const v8h*)(p + 16);
    v16h r;
#pragma unroll
    for (int j = 0; j < 8; ++j) { r[j] = lo[j]; r[8 + j] = hi[j]; }
    return r;
}

__device__ __forceinline__ v16h frag_f32(const float* p) {
    const v4f a0 = *(const v4f*)p,        a1 = *(const v4f*)(p + 4);
    const v4f b0 = *(const v4f*)(p + 16), b1 = *(const v4f*)(p + 20);
    v16h r;
#pragma unroll
    for (int j = 0; j < 4; ++j) {
        r[j]      = (half_t)a0[j];
        r[4 + j]  = (half_t)a1[j];
        r[8 + j]  = (half_t)b0[j];
        r[12 + j] = (half_t)b1[j];
    }
    return r;
}

// ---------------------------------------------------------------------------
// Projection GEMM: C(4096x1024) = X(4096x1024) @ W^T + bias
// MODE 0: q -> f16 [bh][t][64], scaled by 1/8     MODE 1: k -> f16 [bh][t][64]
// MODE 2: v -> f16 [bh][64][t] (transposed)       MODE 3: out -> f32 row-major
// Each wave computes a 16(M) x 64(N) tile; 8 waves / block.
// ---------------------------------------------------------------------------
template <int MODE>
__global__ void __launch_bounds__(256)
proj_kernel(const float* __restrict__ X, const float* __restrict__ W,
            const float* __restrict__ bias, void* __restrict__ out) {
    const int lane  = threadIdx.x & 31;
    const int wid   = threadIdx.x >> 5;
    const int task  = blockIdx.x * 8 + wid;   // 0..4095
    const int mtile = task & 255;             // 256 tiles of 16 rows
    const int nstr  = task >> 8;              // 16 strips of 64 cols
    const int n  = lane & 15;
    const int hi = lane >> 4;

    const float* xrow = X + (size_t)(mtile * 16 + n) * EMBED + 8 * hi;
    const int ocol0 = nstr * 64;

    v8f acc[4] = {v8f{}, v8f{}, v8f{}, v8f{}};
    for (int e0 = 0; e0 < EMBED; e0 += 32) {
        const v16h a = frag_f32(xrow + e0);
#pragma unroll
        for (int c = 0; c < 4; ++c) {
            const int o = ocol0 + c * 16 + n;
            const v16h b = frag_f32(W + (size_t)o * EMBED + e0 + 8 * hi);
            acc[c] = wmma_f16(a, b, acc[c]);
        }
    }

#pragma unroll
    for (int c = 0; c < 4; ++c) {
        const int o  = ocol0 + c * 16 + n;
        const float bv = bias[o];
#pragma unroll
        for (int i = 0; i < 8; ++i) {
            const int r = mtile * 16 + 8 * hi + i;   // C layout: M = 8*hi + i
            float v = acc[c][i] + bv;
            if (MODE == 0) v *= 0.125f;              // HEAD_DIM^-0.5
            if (MODE == 3) {
                // final output: never re-read on device -> non-temporal
                __builtin_nontemporal_store(v, (float*)out + (size_t)r * EMBED + o);
            } else {
                const int t = r >> 1, bb = r & 1;
                const int h = o >> 6, d = o & 63;
                const int bh = bb * NHEADS + h;
                if (MODE == 2)
                    ((half_t*)out)[((size_t)bh * HDIM + d) * SEQ + t] = (half_t)v;
                else
                    ((half_t*)out)[((size_t)bh * SEQ + t) * HDIM + d] = (half_t)v;
            }
        }
    }
}

// ---------------------------------------------------------------------------
// Attention: per block = (bh, 16-row q tile).
// Phase 1: scores S = Q K^T (WMMA, mask applied) -> LDS (16 x 2048, padded)
// Phase 2: row softmax (4-wide vectorized), NT write of attn_weights
// Phase 3: ctx = S @ V via WMMA with split-K across waves + LDS reduce
// ---------------------------------------------------------------------------
__global__ void __launch_bounds__(256)
attn_kernel(const half_t* __restrict__ qh, const half_t* __restrict__ kh,
            const half_t* __restrict__ vt, const unsigned char* __restrict__ mask,
            float* __restrict__ attn_w, float* __restrict__ ctx) {
    extern __shared__ float smem[];
    float* sc  = smem;                 // [16][SROW]
    float* red = smem + 16 * SROW;     // [4][16][16] split-K partials

    const int lane  = threadIdx.x & 31;
    const int wid   = threadIdx.x >> 5;
    const int n     = lane & 15;
    const int hi    = lane >> 4;
    const int qtile = blockIdx.x;      // 0..127
    const int bh    = blockIdx.y;      // 0..31
    const int b     = bh >> 4;
    const int h     = bh & 15;

    // ---- Q fragments (16 rows x 64) held in registers, shared by all chunks
    const half_t* qbase = qh + ((size_t)bh * SEQ + qtile * 16) * HDIM;
    v16h aq[2];
#pragma unroll
    for (int s = 0; s < 2; ++s)
        aq[s] = frag_f16(qbase + (size_t)n * HDIM + 32 * s + 8 * hi);

    // ---- Phase 1: scores
    const half_t* kbase = kh + (size_t)bh * SEQ * HDIM;
    for (int ci = 0; ci < 16; ++ci) {
        const int kc   = wid + 8 * ci;              // k chunk of 16 cols
        const int krow = kc * 16 + n;
        const half_t* kp = kbase + (size_t)krow * HDIM + 8 * hi;
        if (ci < 15) __builtin_prefetch(kp + 128 * HDIM, 0, 1);
        v8f acc = v8f{};
#pragma unroll
        for (int s = 0; s < 2; ++s)
            acc = wmma_f16(aq[s], frag_f16(kp + 32 * s), acc);

        const int col = kc * 16 + n;
        const bool msk = mask[b * SEQ + col] != 0;
#pragma unroll
        for (int i = 0; i < 8; ++i)
            sc[(8 * hi + i) * SROW + col] = msk ? FMIN : acc[i];
    }
    __syncthreads();

    // ---- Phase 2: softmax over each of 16 rows (2 rows per wave), 4-wide
#pragma unroll
    for (int rr = 0; rr < 2; ++rr) {
        const int rowi = wid * 2 + rr;
        float* srow = sc + rowi * SROW;

        float mx = FMIN;
        for (int j = lane * 4; j < SEQ; j += 128) {
            const v4f x = *(const v4f*)(srow + j);
#pragma unroll
            for (int u = 0; u < 4; ++u) mx = fmaxf(mx, x[u]);
        }
#pragma unroll
        for (int off = 16; off > 0; off >>= 1)
            mx = fmaxf(mx, __shfl_xor(mx, off, 32));

        float sum = 0.f;
        for (int j = lane * 4; j < SEQ; j += 128) {
            v4f x = *(const v4f*)(srow + j);
            v4f e;
#pragma unroll
            for (int u = 0; u < 4; ++u) { e[u] = __expf(x[u] - mx); sum += e[u]; }
            *(v4f*)(srow + j) = e;
        }
#pragma unroll
        for (int off = 16; off > 0; off >>= 1)
            sum += __shfl_xor(sum, off, 32);
        const float inv = 1.0f / sum;

        float* gout = attn_w + ((size_t)bh * SEQ + qtile * 16 + rowi) * SEQ;
        for (int j = lane * 4; j < SEQ; j += 128) {
            v4f w = *(const v4f*)(srow + j);
#pragma unroll
            for (int u = 0; u < 4; ++u) w[u] *= inv;
            *(v4f*)(srow + j) = w;
            // 537 MB stream nobody re-reads: NT keeps it out of L2
            __builtin_nontemporal_store(w, (v4f*)(gout + j));
        }
    }
    __syncthreads();

    // ---- Phase 3: ctx(16x64) = S(16x2048) @ V(2048x64), split-K in halves
    const int c     = wid & 3;     // 16-col output chunk
    const int khalf = wid >> 2;    // k-range half
    const float*  srow = sc + (size_t)n * SROW;                 // A rows
    const half_t* vrow = vt + ((size_t)bh * HDIM + c * 16 + n) * SEQ;  // B rows
    v8f acc = v8f{};
    const int ebeg = khalf * 1024;
    for (int e0 = ebeg; e0 < ebeg + 1024; e0 += 32) {
        const v16h a  = frag_f32(srow + e0 + 8 * hi);
        const v16h bb = frag_f16(vrow + e0 + 8 * hi);
        acc = wmma_f16(a, bb, acc);
    }
    if (khalf == 1) {
#pragma unroll
        for (int i = 0; i < 8; ++i)
            red[c * 256 + (8 * hi + i) * 16 + n] = acc[i];
    }
    __syncthreads();
    if (khalf == 0) {
#pragma unroll
        for (int i = 0; i < 8; ++i) {
            const float v = acc[i] + red[c * 256 + (8 * hi + i) * 16 + n];
            const int t = qtile * 16 + 8 * hi + i;
            // ctx row order (t, b) matches the final reshape
            ctx[((size_t)t * BSZ + b) * EMBED + h * HDIM + c * 16 + n] = v;
        }
    }
}

extern "C" void kernel_launch(void* const* d_in, const int* in_sizes, int n_in,
                              void* d_out, int out_size, void* d_ws, size_t ws_size,
                              hipStream_t stream) {
    const float* query = (const float*)d_in[0];
    const unsigned char* mask = (const unsigned char*)d_in[1];  // jnp.bool_ = 1 byte
    const float* qw = (const float*)d_in[2];
    const float* qb = (const float*)d_in[3];
    const float* kw = (const float*)d_in[4];
    const float* kb = (const float*)d_in[5];
    const float* vw = (const float*)d_in[6];
    const float* vb = (const float*)d_in[7];
    const float* ow = (const float*)d_in[8];
    const float* ob = (const float*)d_in[9];

    float* out_attn = (float*)d_out;                              // 2048*2*1024
    float* out_w    = out_attn + (size_t)SEQ * BSZ * EMBED;       // 32*2048*2048

    char* ws = (char*)d_ws;
    half_t* q_h = (half_t*)(ws);                          //  8 MB  [bh][t][64]
    half_t* k_h = (half_t*)(ws + ((size_t)8  << 20));     //  8 MB  [bh][t][64]
    half_t* v_t = (half_t*)(ws + ((size_t)16 << 20));     //  8 MB  [bh][64][t]
    float*  ctx = (float*) (ws + ((size_t)24 << 20));     // 16 MB  [t*2+b][1024]

    proj_kernel<0><<<512, 256, 0, stream>>>(query, qw, qb, (void*)q_h);
    proj_kernel<1><<<512, 256, 0, stream>>>(query, kw, kb, (void*)k_h);
    proj_kernel<2><<<512, 256, 0, stream>>>(query, vw, vb, (void*)v_t);

    const size_t lds_bytes = (16 * SROW + 4 * 256) * sizeof(float); // ~135 KB (<320 KB/WGP)
    attn_kernel<<<dim3(128, 32), 256, lds_bytes, stream>>>(q_h, k_h, v_t, mask,
                                                           out_w, ctx);

    proj_kernel<3><<<512, 256, 0, stream>>>(ctx, ow, ob, (void*)out_attn);
}
